// Attention_6786048328298
// MI455X (gfx1250) — matrix operand order repaired and checked by reading, not compiled, not run on hardware
//
#include <hip/hip_runtime.h>

typedef __attribute__((ext_vector_type(16))) _Float16 v16h;
typedef __attribute__((ext_vector_type(8)))  _Float16 v8h;
typedef __attribute__((ext_vector_type(4)))  _Float16 v4h;
typedef __attribute__((ext_vector_type(8)))  float    v8f;
typedef __attribute__((ext_vector_type(4)))  float    v4f;

#define S_LEN 2048
#define D_DIM 128
#define BM    128           // q rows per workgroup (8 waves x 16)
#define BN    32            // keys per iteration
#define NW    8
#define KSTR  (D_DIM + 16)  // 144 halfs -> 288 B rows (32B aligned for v16h)
#define VSTR  40            // transposed V rows: 80 B   (16B aligned for v8h)
#define PSTR  40            // 16x32 P tile + pad, per wave
#define ATTN_SCALE 0.08838834764831845f  // 1/sqrt(128)

__global__ __launch_bounds__(256)
void fa_fwd_kernel(const float* __restrict__ Q, const float* __restrict__ K,
                   const float* __restrict__ V, float* __restrict__ O) {
  __shared__ _Float16 Ksh[2][BN * KSTR];      // double buffered
  __shared__ _Float16 Vsh[2][D_DIM * VSTR];   // transposed: [d][key]
  __shared__ _Float16 Psh[NW * 16 * PSTR];

  const int tid  = threadIdx.x;
  const int wave = tid >> 5;
  const int lane = tid & 31;
  const int l16  = lane & 15;
  const int hi16 = lane >> 4;                 // which 16-lane half

  const int q0 = blockIdx.x * BM;
  const size_t base = ((size_t)(blockIdx.z * 16 + blockIdx.y)) * S_LEN * D_DIM;
  const float* Qg = Q + base;
  const float* Kg = K + base;
  const float* Vg = V + base;
  float*       Og = O + base;

  // ---- staging micro-tile: each thread owns a 4(key) x 4(d) block ----
  const int kt = (tid >> 5) << 2;             // key row group 0..28
  const int dt = (tid & 31) << 2;             // d column group 0..124

  v4f kreg[4], vreg[4];
  auto issue_loads = [&](int kbase) {
#pragma unroll
    for (int j = 0; j < 4; ++j) {
      const size_t goff = (size_t)(kbase + kt + j) * D_DIM + dt;
      kreg[j] = *(const v4f*)(Kg + goff);
      vreg[j] = *(const v4f*)(Vg + goff);
    }
  };
  auto store_lds = [&](int buf) {
#pragma unroll
    for (int j = 0; j < 4; ++j) {             // K: row-major f16
      v4h kh;
      kh[0] = (_Float16)kreg[j][0]; kh[1] = (_Float16)kreg[j][1];
      kh[2] = (_Float16)kreg[j][2]; kh[3] = (_Float16)kreg[j][3];
      *(v4h*)&Ksh[buf][(kt + j) * KSTR + dt] = kh;
    }
#pragma unroll
    for (int i = 0; i < 4; ++i) {             // V: 4x4 register transpose
      v4h vh;
      vh[0] = (_Float16)vreg[0][i]; vh[1] = (_Float16)vreg[1][i];
      vh[2] = (_Float16)vreg[2][i]; vh[3] = (_Float16)vreg[3][i];
      *(v4h*)&Vsh[buf][(dt + i) * VSTR + kt] = vh;
    }
  };

  // ---- load Q fragments once (1/sqrt(d) folded in), WMMA 16-bit A layout ----
  const int qrow = q0 + wave * 16 + l16;
  v16h aq[4];
#pragma unroll
  for (int c = 0; c < 4; ++c) {
    const float* p0 = Qg + (size_t)qrow * D_DIM + c * 32 + hi16 * 16;
    const float* p1 = p0 + 8;
    v4f x0 = *(const v4f*)(p0);
    v4f x1 = *(const v4f*)(p0 + 4);
    v4f y0 = *(const v4f*)(p1);
    v4f y1 = *(const v4f*)(p1 + 4);
#pragma unroll
    for (int e = 0; e < 4; ++e) {
      aq[c][e]      = (_Float16)(x0[e] * ATTN_SCALE);
      aq[c][4 + e]  = (_Float16)(x1[e] * ATTN_SCALE);
      aq[c][8 + e]  = (_Float16)(y0[e] * ATTN_SCALE);
      aq[c][12 + e] = (_Float16)(y1[e] * ATTN_SCALE);
    }
  }

  // ---- softmax state (row = r + 8*hi16 within the wave's 16-row strip) ----
  float mrow[8], lrow[8];
  v8f   oacc[8];
#pragma unroll
  for (int r = 0; r < 8; ++r) { mrow[r] = -__builtin_inff(); lrow[r] = 0.0f; }
#pragma unroll
  for (int td = 0; td < 8; ++td)
#pragma unroll
    for (int r = 0; r < 8; ++r) oacc[td][r] = 0.0f;

  const int wq0 = q0 + wave * 16;             // wave's first q row
  const int nkb = (q0 + BM) / BN;             // causal loop bound

  // ---- prologue: fill buffer 0 ----
  issue_loads(0);
  store_lds(0);
  __syncthreads();

  for (int kb = 0; kb < nkb; ++kb) {
    const int kbase = kb * BN;
    const int cur   = kb & 1;
    const bool havenext = (kb + 1) < nkb;
    if (havenext) issue_loads(kbase + BN);    // prefetch next block into regs

    if (kbase <= wq0 + 15) {                  // wave-uniform causal skip
      // ---- S = Q K^T : two 16x16 tiles, K-dim 128 in 4 chunks ----
      v8f st[2];
#pragma unroll
      for (int t = 0; t < 2; ++t)
#pragma unroll
        for (int r = 0; r < 8; ++r) st[t][r] = 0.0f;
#pragma unroll
      for (int t = 0; t < 2; ++t)
#pragma unroll
        for (int c = 0; c < 4; ++c) {
          const v16h bk = *(const v16h*)
              &Ksh[cur][(t * 16 + l16) * KSTR + c * 32 + hi16 * 16];
          st[t] = __builtin_amdgcn_wmma_f32_16x16x32_f16(
              false, aq[c], false, bk, (short)0, st[t], false, false);
        }

      // ---- causal mask (uniform skip when block is fully unmasked) ----
      if (kbase + BN - 1 > wq0) {
#pragma unroll
        for (int t = 0; t < 2; ++t) {
          const int key = kbase + t * 16 + l16;
#pragma unroll
          for (int r = 0; r < 8; ++r)
            if (key > (wq0 + r + 8 * hi16)) st[t][r] = -__builtin_inff();
        }
      }

      // ---- online softmax: row max / rescale / row sum ----
      float mt[8];
#pragma unroll
      for (int r = 0; r < 8; ++r) mt[r] = fmaxf(st[0][r], st[1][r]);
#pragma unroll
      for (int off = 1; off < 16; off <<= 1)
#pragma unroll
        for (int r = 0; r < 8; ++r)
          mt[r] = fmaxf(mt[r], __shfl_xor(mt[r], off, 32));

      float corr[8];
#pragma unroll
      for (int r = 0; r < 8; ++r) {
        const float mn = fmaxf(mrow[r], mt[r]);
        corr[r] = __expf(mrow[r] - mn);
        mrow[r] = mn;
      }
#pragma unroll
      for (int t = 0; t < 2; ++t)
#pragma unroll
        for (int r = 0; r < 8; ++r) st[t][r] = __expf(st[t][r] - mrow[r]);

      float rs[8];
#pragma unroll
      for (int r = 0; r < 8; ++r) rs[r] = st[0][r] + st[1][r];
#pragma unroll
      for (int off = 1; off < 16; off <<= 1)
#pragma unroll
        for (int r = 0; r < 8; ++r) rs[r] += __shfl_xor(rs[r], off, 32);
#pragma unroll
      for (int r = 0; r < 8; ++r) lrow[r] = lrow[r] * corr[r] + rs[r];
#pragma unroll
      for (int td = 0; td < 8; ++td)
#pragma unroll
        for (int r = 0; r < 8; ++r) oacc[td][r] *= corr[r];

      // ---- P: C/D layout -> A layout via per-wave LDS scratch ----
      _Float16* pw = &Psh[wave * 16 * PSTR];
#pragma unroll
      for (int t = 0; t < 2; ++t)
#pragma unroll
        for (int r = 0; r < 8; ++r)
          pw[(r + 8 * hi16) * PSTR + t * 16 + l16] = (_Float16)st[t][r];
      __builtin_amdgcn_fence(__ATOMIC_ACQ_REL, "wavefront");
      __builtin_amdgcn_wave_barrier();        // DS is in-order per wave

      v16h ap;
      {
        const v8h plo = *(const v8h*)&pw[l16 * PSTR + hi16 * 8];
        const v8h phi = *(const v8h*)&pw[l16 * PSTR + 16 + hi16 * 8];
#pragma unroll
        for (int e = 0; e < 8; ++e) { ap[e] = plo[e]; ap[8 + e] = phi[e]; }
      }

      // ---- O += P V : 8 d-tiles, single K=32 chunk each ----
#pragma unroll
      for (int td = 0; td < 8; ++td) {
        const v8h b0 = *(const v8h*)
            &Vsh[cur][(td * 16 + l16) * VSTR + hi16 * 8];
        const v8h b1 = *(const v8h*)
            &Vsh[cur][(td * 16 + l16) * VSTR + 16 + hi16 * 8];
        v16h bv;
#pragma unroll
        for (int e = 0; e < 8; ++e) { bv[e] = b0[e]; bv[8 + e] = b1[e]; }
        oacc[td] = __builtin_amdgcn_wmma_f32_16x16x32_f16(
            false, ap, false, bv, (short)0, oacc[td], false, false);
      }
    }

    if (havenext) store_lds(cur ^ 1);         // fill the other buffer
    __syncthreads();
  }

  // ---- epilogue: normalize and store fp32 ----
#pragma unroll
  for (int r = 0; r < 8; ++r) {
    const float inv = 1.0f / lrow[r];
    const int row = wq0 + r + 8 * hi16;
#pragma unroll
    for (int td = 0; td < 8; ++td)
      Og[(size_t)row * D_DIM + td * 16 + l16] = oacc[td][r] * inv;
  }
}

extern "C" void kernel_launch(void* const* d_in, const int* in_sizes, int n_in,
                              void* d_out, int out_size, void* d_ws, size_t ws_size,
                              hipStream_t stream) {
  (void)in_sizes; (void)n_in; (void)d_ws; (void)ws_size; (void)out_size;
  const float* Q = (const float*)d_in[0];
  const float* K = (const float*)d_in[1];
  const float* V = (const float*)d_in[2];
  // d_in[3] is the causal mask; causality is computed analytically.
  float* O = (float*)d_out;
  dim3 grid(S_LEN / BM, 16, 4);   // (q-tiles, H, B)
  fa_fwd_kernel<<<grid, dim3(256), 0, stream>>>(Q, K, V, O);
}
